// MultiHeadAttention_34316788695287
// MI455X (gfx1250) — matrix operand order repaired and checked by reading, not compiled, not run on hardware
//
#include <hip/hip_runtime.h>
#include <hip/hip_bf16.h>
#include <stdint.h>

typedef __attribute__((ext_vector_type(16))) _Float16 v16h;
typedef __attribute__((ext_vector_type(8)))  float    v8f;

#define TILE_M 128
#define TILE_N 64
#define TILE_K 32
#define NTH    128   // 4 waves (wave32); each wave owns 32 output rows

// ---- CDNA5 async global->LDS copy (ASYNCcnt-tracked), 16 bytes per lane ----
__device__ __forceinline__ void async_copy_b128(uint32_t lds_off, const void* gptr) {
    asm volatile("global_load_async_to_lds_b128 %0, %1, off"
                 :: "v"(lds_off), "v"(gptr) : "memory");
}
__device__ __forceinline__ void async_wait0() {
    asm volatile("s_wait_asynccnt 0x0" ::: "memory");
}
__device__ __forceinline__ uint32_t lds_off_of(const void* p) {
    return (uint32_t)(uintptr_t)p;   // low 32 bits of flat LDS address = LDS offset
}

// Generic batched WMMA GEMM: C[z] = alpha * A[z](MxK) * B[z](KxN) (+ bias)
//  - A row-major [m][k]; f16 sources staged with async-to-LDS, f32 converted
//  - B_NMAJOR=true : B memory is [n][k] row-major (K rows / E rows)
//    B_NMAJOR=false: B memory is [k][n] row-major (weights, V) -> LDS transpose
//  - per-z offsets: (z / zdiv) * outer + (z % zdiv) * inner
template<typename AT, typename BT, bool B_NMAJOR, bool HAS_BIAS, bool OUT_HALF>
__global__ __launch_bounds__(NTH)
void wmma_gemm(const AT* __restrict__ Ag, long lda, long aInner, long aOuter,
               const BT* __restrict__ Bg, long ldb, long bInner, long bOuter,
               const float* __restrict__ bias,
               void* __restrict__ Cg, long ldc, long cInner, long cOuter,
               int M, int N, int K, int zdiv, float alpha)
{
    constexpr bool A_HALF = (sizeof(AT) == 2);
    constexpr bool B_HALF = (sizeof(BT) == 2);
    constexpr bool ASYNC_A = A_HALF;
    constexpr bool ASYNC_B = (B_HALF && B_NMAJOR);

    __shared__ _Float16 sA[TILE_M * TILE_K];   // [m][k]  8KB
    __shared__ _Float16 sB[TILE_N * TILE_K];   // [n][k]  4KB

    const int  z  = blockIdx.z;
    const long zo = (long)(z / zdiv);
    const long zi = (long)(z % zdiv);
    const AT* A = Ag + zo * aOuter + zi * aInner;
    const BT* B = Bg + zo * bOuter + zi * bInner;

    const int m0   = blockIdx.y * TILE_M;
    const int n0   = blockIdx.x * TILE_N;
    const int tid  = threadIdx.x;
    const int wid  = tid >> 5;
    const int lane = tid & 31;
    const int lr   = lane & 15;   // row (A) / col (B,C) selector
    const int lg   = lane >> 4;   // half-wave group

    v8f acc[2][4];
    #pragma unroll
    for (int mf = 0; mf < 2; ++mf)
        #pragma unroll
        for (int t = 0; t < 4; ++t)
            #pragma unroll
            for (int e = 0; e < 8; ++e) acc[mf][t][e] = 0.0f;

    for (int k0 = 0; k0 < K; k0 += TILE_K) {
        // ---- stage A tile: 512 chunks of 8 halves (M,K tiles always full) ----
        for (int c = tid; c < (TILE_M * TILE_K) / 8; c += NTH) {
            const int row = c >> 2, col = (c & 3) * 8;
            const AT* src = A + (long)(m0 + row) * lda + k0 + col;
            _Float16* dst = &sA[row * TILE_K + col];
            if constexpr (ASYNC_A) {
                async_copy_b128(lds_off_of(dst), src);
            } else {
                #pragma unroll
                for (int e = 0; e < 8; ++e) dst[e] = (_Float16)(float)src[e];
            }
        }
        // ---- stage B tile into [n][k] ----
        if constexpr (B_NMAJOR) {
            for (int c = tid; c < (TILE_N * TILE_K) / 8; c += NTH) {
                const int row = c >> 2, col = (c & 3) * 8;
                const int gn = n0 + row;
                _Float16* dst = &sB[row * TILE_K + col];
                if (gn < N) {
                    const BT* src = B + (long)gn * ldb + k0 + col;
                    if constexpr (ASYNC_B) {
                        async_copy_b128(lds_off_of(dst), src);
                    } else {
                        #pragma unroll
                        for (int e = 0; e < 8; ++e) dst[e] = (_Float16)(float)src[e];
                    }
                } else {
                    #pragma unroll
                    for (int e = 0; e < 8; ++e) dst[e] = (_Float16)0.0f;
                }
            }
        } else {
            for (int idx = tid; idx < TILE_N * TILE_K; idx += NTH) {
                const int kk = idx >> 6;      // 0..31
                const int nn = idx & 63;      // 0..63
                const int gn = n0 + nn;
                float v = (gn < N) ? (float)B[(long)(k0 + kk) * ldb + gn] : 0.0f;
                sB[nn * TILE_K + kk] = (_Float16)v;
            }
        }
        if constexpr (ASYNC_A || ASYNC_B) async_wait0();
        __syncthreads();

        // ---- 2 A fragments per wave (rows wid*32 + mf*16), ISA 16x32 layout ----
        v16h afrag[2];
        #pragma unroll
        for (int mf = 0; mf < 2; ++mf) {
            const _Float16* ap = &sA[(wid * 32 + mf * 16 + lr) * TILE_K + 8 * lg];
            #pragma unroll
            for (int e = 0; e < 8; ++e) { afrag[mf][e] = ap[e]; afrag[mf][8 + e] = ap[16 + e]; }
        }
        // ---- 4 B fragments, each reused by both A fragments: 8 WMMAs ----
        #pragma unroll
        for (int nt = 0; nt < 4; ++nt) {
            v16h bfrag;
            const _Float16* bp = &sB[(nt * 16 + lr) * TILE_K + 8 * lg];
            #pragma unroll
            for (int e = 0; e < 16; ++e) bfrag[e] = bp[e + (e & 8)];
            #pragma unroll
            for (int mf = 0; mf < 2; ++mf)
                acc[mf][nt] = __builtin_amdgcn_wmma_f32_16x16x32_f16(
                    false, afrag[mf], false, bfrag, (short)0, acc[mf][nt], false, false);
        }
        __syncthreads();
    }

    // ---- epilogue (C/D layout: col = lr, row r + 8*lg) ----
    const long zc = zo * cOuter + zi * cInner;
    #pragma unroll
    for (int mf = 0; mf < 2; ++mf) {
        #pragma unroll
        for (int nt = 0; nt < 4; ++nt) {
            const int col = n0 + nt * 16 + lr;
            if (col >= N) continue;
            float bv = 0.0f;
            if constexpr (HAS_BIAS) bv = bias[col];
            #pragma unroll
            for (int r = 0; r < 8; ++r) {
                const int row = m0 + wid * 32 + mf * 16 + r + 8 * lg;
                const float val = acc[mf][nt][r] * alpha + bv;
                if constexpr (OUT_HALF)
                    ((_Float16*)Cg)[zc + (long)row * ldc + col] = (_Float16)val;
                else
                    ((float*)Cg)[zc + (long)row * ldc + col] = val;
            }
        }
    }
}

// Softmax over each (b,h,i) row of length 1024, fusing the rel-shift bias
// gather from Y = Q * E^T (padded to 2048 cols). Closed form of
// _rel_shift(gathered q.rel):
//   j <= i   : Y[i, 2i - j]
//   j == i+1 : 0
//   j >= i+2 : Y[i+1, 1026 + 2i - j]      (1026 = S + 2)
__global__ __launch_bounds__(256)
void softmax_bias_kernel(const float* __restrict__ scores,
                         const _Float16* __restrict__ Y,
                         _Float16* __restrict__ P)
{
    const int S = 1024;
    const int row = blockIdx.x;          // b*H*S + h*S + i
    const int bh  = row >> 10;
    const int i   = row & 1023;
    const float* srow = scores + (size_t)row * S;
    const _Float16* Yb = Y + (size_t)bh * S * 2048;
    _Float16* prow = P + (size_t)row * S;
    const int tid = threadIdx.x;

    float x[4];
    #pragma unroll
    for (int k = 0; k < 4; ++k) {
        const int j = tid + k * 256;
        float b;
        if (j <= i)            b = (float)Yb[(size_t)i * 2048 + (2 * i - j)];
        else if (j == i + 1)   b = 0.0f;
        else                   b = (float)Yb[(size_t)(i + 1) * 2048 + (1026 + 2 * i - j)];
        x[k] = srow[j] + b;    // scores already scaled by 1/sqrt(KD)
    }

    __shared__ float red[8];
    // --- row max ---
    float m = fmaxf(fmaxf(x[0], x[1]), fmaxf(x[2], x[3]));
    #pragma unroll
    for (int off = 16; off > 0; off >>= 1) m = fmaxf(m, __shfl_xor(m, off, 32));
    if ((tid & 31) == 0) red[tid >> 5] = m;
    __syncthreads();
    float gm = red[0];
    #pragma unroll
    for (int w = 1; w < 8; ++w) gm = fmaxf(gm, red[w]);
    __syncthreads();

    // --- exp + row sum ---
    float e[4], s = 0.0f;
    #pragma unroll
    for (int k = 0; k < 4; ++k) { e[k] = __expf(x[k] - gm); s += e[k]; }
    #pragma unroll
    for (int off = 16; off > 0; off >>= 1) s += __shfl_xor(s, off, 32);
    if ((tid & 31) == 0) red[tid >> 5] = s;
    __syncthreads();
    float gs = red[0];
    #pragma unroll
    for (int w = 1; w < 8; ++w) gs += red[w];
    const float inv = 1.0f / gs;

    #pragma unroll
    for (int k = 0; k < 4; ++k)
        prow[tid + k * 256] = (_Float16)(e[k] * inv);
}

extern "C" void kernel_launch(void* const* d_in, const int* in_sizes, int n_in,
                              void* d_out, int out_size, void* d_ws, size_t ws_size,
                              hipStream_t stream)
{
    const float* q_in = (const float*)d_in[0];
    const float* k_in = (const float*)d_in[1];
    const float* v_in = (const float*)d_in[2];
    const float* Wq   = (const float*)d_in[3];
    const float* bq   = (const float*)d_in[4];
    const float* Wk   = (const float*)d_in[5];
    const float* bk   = (const float*)d_in[6];
    const float* Wv   = (const float*)d_in[7];
    const float* bv   = (const float*)d_in[8];
    const float* Wo   = (const float*)d_in[9];
    const float* bo   = (const float*)d_in[10];
    const float* rel  = (const float*)d_in[11];

    const int  Bc = 2, S = 1024, D = 1024, H = 16, KD = 64;
    const long MS = (long)Bc * S;              // 2048
    const long SD = (long)S * D;               // per-batch activation stride

    // workspace carve-out (~352 MB)
    char* ws = (char*)d_ws;
    size_t off = 0;
    auto carve = [&](size_t bytes) {
        void* p = ws + off;
        off = (off + bytes + 255) & ~(size_t)255;
        return p;
    };
    _Float16* Qp = (_Float16*)carve(MS * D * sizeof(_Float16));              // 4 MB
    _Float16* Kp = (_Float16*)carve(MS * D * sizeof(_Float16));              // 4 MB
    _Float16* Vp = (_Float16*)carve(MS * D * sizeof(_Float16));              // 4 MB
    _Float16* Oh = (_Float16*)carve(MS * D * sizeof(_Float16));              // 4 MB
    _Float16* Y  = (_Float16*)carve((size_t)Bc * H * S * 2048 * sizeof(_Float16)); // 134 MB
    _Float16* P  = (_Float16*)carve((size_t)Bc * H * S * S * sizeof(_Float16));    // 67 MB
    float*  Sc   = (float*)   carve((size_t)Bc * H * S * S * sizeof(float));       // 134 MB

    const dim3 blk(NTH);

    // 1) Q/K/V projections: f32 A x f32 W -> f16, +bias
    const dim3 gproj(D / TILE_N, MS / TILE_M, 1);
    wmma_gemm<float, float, false, true, true><<<gproj, blk, 0, stream>>>(
        q_in, D, 0, 0, Wq, D, 0, 0, bq, Qp, D, 0, 0, (int)MS, D, D, 1, 1.0f);
    wmma_gemm<float, float, false, true, true><<<gproj, blk, 0, stream>>>(
        k_in, D, 0, 0, Wk, D, 0, 0, bk, Kp, D, 0, 0, (int)MS, D, D, 1, 1.0f);
    wmma_gemm<float, float, false, true, true><<<gproj, blk, 0, stream>>>(
        v_in, D, 0, 0, Wv, D, 0, 0, bv, Vp, D, 0, 0, (int)MS, D, D, 1, 1.0f);

    // 2) scores = (Q_h . K_h^T) / 8 : per (b,h), f16 inputs (async staged), f32 out
    const dim3 gsc(S / TILE_N, S / TILE_M, Bc * H);
    wmma_gemm<_Float16, _Float16, true, false, false><<<gsc, blk, 0, stream>>>(
        Qp, D, KD, SD, Kp, D, KD, SD, nullptr,
        Sc, S, (long)S * S, (long)H * S * S, S, S, KD, H, 0.125f);

    // 3) Y = Q_h . E^T : per (b,h), N = 2S-1 (padded to 2048), f16 out
    const dim3 grel(2048 / TILE_N, S / TILE_M, Bc * H);
    wmma_gemm<_Float16, float, true, false, true><<<grel, blk, 0, stream>>>(
        Qp, D, KD, SD, rel, KD, 0, 0, nullptr,
        Y, 2048, (long)S * 2048, (long)H * S * 2048, S, 2 * S - 1, KD, H, 1.0f);

    // 4) softmax(scores + rel_shift_bias) -> P (f16)
    softmax_bias_kernel<<<Bc * H * S, 256, 0, stream>>>(Sc, Y, P);

    // 5) O_h = P . V_h : per (b,h), N = 64, V transposed on LDS store
    const dim3 gav(1, S / TILE_M, Bc * H);
    wmma_gemm<_Float16, _Float16, false, false, true><<<gav, blk, 0, stream>>>(
        P, S, (long)S * S, (long)H * S * S, Vp, D, KD, SD, nullptr,
        Oh, D, KD, SD, S, KD, S, H, 1.0f);

    // 6) out = O . Wo + bo : f32 output
    const dim3 gfin(D / TILE_N, MS / TILE_M, 1);
    wmma_gemm<_Float16, float, false, true, false><<<gfin, blk, 0, stream>>>(
        Oh, D, 0, 0, Wo, D, 0, 0, bo, d_out, D, 0, 0, (int)MS, D, D, 1, 1.0f);
}
